// DrugRepurposingGNN_15659450761599
// MI455X (gfx1250) — compile-verified, matrix operand order checked by reading, not run on hardware
//
#include <hip/hip_runtime.h>
#include <hip/hip_bf16.h>
#include <hip/hip_fp16.h>

#define NN  50000
#define NE  320000
#define RR  10
#define EMB 128
#define HID 256
#define SS  3
#define NH  4
#define HD  64

typedef __attribute__((ext_vector_type(16))) _Float16 v16h;
typedef __attribute__((ext_vector_type(8)))  float    v8f;

union Frag16 { v16h v; uint4 u[2]; };

__device__ __forceinline__ void load_frag(Frag16& f, const _Float16* p) {
  f.u[0] = *(const uint4*)(p);        // 8 contiguous halves
  f.u[1] = *(const uint4*)(p + 16);   // next 8 (K+16..23 per lane group)
}

__device__ __forceinline__ float wsum(float v) {
#pragma unroll
  for (int off = 16; off > 0; off >>= 1) v += __shfl_xor(v, off, 32);
  return v;
}

// ---------------- weight transpose + f32->f16 ----------------
__global__ void k_wprep(const float* __restrict__ W, _Float16* __restrict__ Wt,
                        int K, int Nc) {
  int idx = blockIdx.x * blockDim.x + threadIdx.x;
  if (idx >= K * Nc) return;
  int k = idx / Nc, n = idx % Nc;
  Wt[(size_t)n * K + k] = (_Float16)W[idx];
}

// ---------------- embedding gather ----------------
__global__ void k_gather(const int* __restrict__ ids, const int* __restrict__ types,
                         const float* __restrict__ nemb, const float* __restrict__ temb,
                         _Float16* __restrict__ x16) {
  int idx = blockIdx.x * blockDim.x + threadIdx.x;
  if (idx >= NN * EMB) return;
  int n = idx >> 7, c = idx & 127;
  float v = nemb[(size_t)ids[n] * EMB + c] + temb[(size_t)types[n] * EMB + c];
  x16[idx] = (_Float16)v;
}

// ---------------- generic f32 -> f16 ----------------
__global__ void k_f2h(const float* __restrict__ in, _Float16* __restrict__ out,
                      long long count) {
  long long i = (long long)blockIdx.x * blockDim.x + threadIdx.x;
  if (i < count) out[i] = (_Float16)in[i];
}

// ---------------- WMMA GEMM: C[M,Nc] = A[M,K](f16) * Bt[Nc,K](f16)^T + bias ----
// Wave tile 16x64 (4 accumulators), K-loop software-pipelined (ping-pong) so
// the 10 loads for step i+1 are in flight while the 4 WMMAs of step i execute.
__global__ __launch_bounds__(256)
void k_gemm(const _Float16* __restrict__ A, const _Float16* __restrict__ Bt,
            const float* __restrict__ bias, float* __restrict__ Cf,
            _Float16* __restrict__ Ch, int M, int K, int Nc) {
  const int lane   = threadIdx.x & 31;
  const int wave   = (blockIdx.x * blockDim.x + threadIdx.x) >> 5;
  const int ntiles = Nc >> 6;              // 64-wide N tiles per wave
  const int mtiles = M >> 4;
  const int mt = wave / ntiles;
  const int nt = wave % ntiles;
  if (mt >= mtiles) return;                // uniform per wave -> EXEC all-ones inside

  const int mrow  = mt * 16 + (lane & 15);
  const int khalf = (lane >> 4) * 8;       // lanes 16-31 hold K+8..15 / +24..31
  const _Float16* arow = A + (size_t)mrow * K + khalf;
  const _Float16* brow[4];
#pragma unroll
  for (int s = 0; s < 4; ++s) {
    int ncol = (nt << 6) + (s << 4) + (lane & 15);
    brow[s] = Bt + (size_t)ncol * K + khalf;
  }

  v8f acc[4];
#pragma unroll
  for (int s = 0; s < 4; ++s)
#pragma unroll
    for (int j = 0; j < 8; ++j) acc[s][j] = 0.0f;

  Frag16 a0, a1, b0[4], b1[4];
  load_frag(a0, arow);
#pragma unroll
  for (int s = 0; s < 4; ++s) load_frag(b0[s], brow[s]);

  // K is always a multiple of 64 here (128 or 256): process K-steps in pairs.
  for (int k0 = 0; k0 < K; k0 += 64) {
    load_frag(a1, arow + k0 + 32);
#pragma unroll
    for (int s = 0; s < 4; ++s) load_frag(b1[s], brow[s] + k0 + 32);
#pragma unroll
    for (int s = 0; s < 4; ++s)
      acc[s] = __builtin_amdgcn_wmma_f32_16x16x32_f16(
          false, a0.v, false, b0[s].v, (short)0, acc[s], false, false);
    if (k0 + 64 < K) {
      load_frag(a0, arow + k0 + 64);
#pragma unroll
      for (int s = 0; s < 4; ++s) load_frag(b0[s], brow[s] + k0 + 64);
    }
#pragma unroll
    for (int s = 0; s < 4; ++s)
      acc[s] = __builtin_amdgcn_wmma_f32_16x16x32_f16(
          false, a1.v, false, b1[s].v, (short)0, acc[s], false, false);
  }

  const int mbase = mt * 16 + ((lane >> 4) << 3);
#pragma unroll
  for (int s = 0; s < 4; ++s) {
    int ncol = (nt << 6) + (s << 4) + (lane & 15);
    float bv = bias ? bias[ncol] : 0.0f;
#pragma unroll
    for (int j = 0; j < 8; ++j) {
      int m = mbase + j;
      float v = acc[s][j] + bv;
      if (Cf) Cf[(size_t)m * Nc + ncol] = v;
      if (Ch) Ch[(size_t)m * Nc + ncol] = (_Float16)v;
    }
  }
}

// ---------------- per-relation edge scatter: out[dst] += h[src] + rel_b ------
__global__ __launch_bounds__(256)
void k_scatter(const int* __restrict__ src, const int* __restrict__ dst,
               const int* __restrict__ et, const float* __restrict__ h,
               const float* __restrict__ relb, int r, float* __restrict__ out) {
  int e = (blockIdx.x * blockDim.x + threadIdx.x) >> 5;
  if (e >= NE) return;
  if (et[e] != r) return;                  // uniform per wave
  int lane = threadIdx.x & 31;
  const float* hs = h + (size_t)src[e] * HID;
  float* od = out + (size_t)dst[e] * HID;
#pragma unroll
  for (int i = 0; i < HID / 32; ++i) {
    int c = lane + 32 * i;
    atomicAdd(&od[c], hs[c] + relb[c]);
  }
}

// ---------------- fused LN1 -> relu -> (+res) -> LN2 ----------------
__global__ __launch_bounds__(256)
void k_postconv(const float* __restrict__ conv, const float* __restrict__ res,
                const float* __restrict__ g1, const float* __restrict__ b1,
                const float* __restrict__ g2, const float* __restrict__ b2,
                float* __restrict__ xout, float* __restrict__ L,
                _Float16* __restrict__ x16, int layer) {
  int n = (blockIdx.x * blockDim.x + threadIdx.x) >> 5;
  if (n >= NN) return;
  int lane = threadIdx.x & 31;
  const float* row = conv + (size_t)n * HID;
  float v[8], s = 0.f;
#pragma unroll
  for (int i = 0; i < 8; ++i) { v[i] = row[lane + 32 * i]; s += v[i]; }
  float mean = wsum(s) * (1.0f / HID);
  float q = 0.f;
#pragma unroll
  for (int i = 0; i < 8; ++i) { float d = v[i] - mean; q += d * d; }
  float rs = rsqrtf(wsum(q) * (1.0f / HID) + 1e-5f);
  s = 0.f;
#pragma unroll
  for (int i = 0; i < 8; ++i) {
    int c = lane + 32 * i;
    float y = (v[i] - mean) * rs * g1[c] + b1[c];
    y = fmaxf(y, 0.0f);
    if (res) y += res[(size_t)n * HID + c];
    v[i] = y; s += y;
  }
  float mean2 = wsum(s) * (1.0f / HID);
  q = 0.f;
#pragma unroll
  for (int i = 0; i < 8; ++i) { float d = v[i] - mean2; q += d * d; }
  float rs2 = rsqrtf(wsum(q) * (1.0f / HID) + 1e-5f);
#pragma unroll
  for (int i = 0; i < 8; ++i) {
    int c = lane + 32 * i;
    float z = (v[i] - mean2) * rs2 * g2[c] + b2[c];
    xout[(size_t)n * HID + c] = z;
    L[((size_t)n * SS + layer) * HID + c] = z;
    x16[(size_t)n * HID + c] = (_Float16)z;
  }
}

// ---------------- fused attention: obar[n] = mean_s(softmax(qk^T/8) @ v) -----
__global__ __launch_bounds__(256)
void k_attn(const _Float16* __restrict__ q, const _Float16* __restrict__ k,
            const _Float16* __restrict__ v, _Float16* __restrict__ obar) {
  int w = (blockIdx.x * blockDim.x + threadIdx.x) >> 5;
  if (w >= NN * NH) return;
  int lane = threadIdx.x & 31;
  int n = w >> 2, h = w & 3;
  size_t base = (size_t)n * SS * HID + h * HD;
  float qv[SS][2], kv[SS][2], vv[SS][2];
#pragma unroll
  for (int s = 0; s < SS; ++s) {
    size_t o = base + (size_t)s * HID + lane * 2;
    qv[s][0] = (float)q[o]; qv[s][1] = (float)q[o + 1];
    kv[s][0] = (float)k[o]; kv[s][1] = (float)k[o + 1];
    vv[s][0] = (float)v[o]; vv[s][1] = (float)v[o + 1];
  }
  float att[SS][SS];
#pragma unroll
  for (int si = 0; si < SS; ++si)
#pragma unroll
    for (int sj = 0; sj < SS; ++sj) {
      float d = qv[si][0] * kv[sj][0] + qv[si][1] * kv[sj][1];
      att[si][sj] = wsum(d) * 0.125f;   // 1/sqrt(64)
    }
#pragma unroll
  for (int si = 0; si < SS; ++si) {
    float m = fmaxf(att[si][0], fmaxf(att[si][1], att[si][2]));
    float e0 = __expf(att[si][0] - m), e1 = __expf(att[si][1] - m), e2 = __expf(att[si][2] - m);
    float inv = 1.0f / (e0 + e1 + e2);
    att[si][0] = e0 * inv; att[si][1] = e1 * inv; att[si][2] = e2 * inv;
  }
  float wc[SS];
#pragma unroll
  for (int sj = 0; sj < SS; ++sj)
    wc[sj] = (att[0][sj] + att[1][sj] + att[2][sj]) * (1.0f / 3.0f);
  float o0 = wc[0] * vv[0][0] + wc[1] * vv[1][0] + wc[2] * vv[2][0];
  float o1 = wc[0] * vv[0][1] + wc[1] * vv[1][1] + wc[2] * vv[2][1];
  size_t ob = (size_t)n * HID + h * HD + lane * 2;
  obar[ob]     = (_Float16)o0;
  obar[ob + 1] = (_Float16)o1;
}

// ---------------- final MLP LN -> relu -> f16 ----------------
__global__ __launch_bounds__(256)
void k_lnrelu(const float* __restrict__ in, const float* __restrict__ g,
              const float* __restrict__ b, _Float16* __restrict__ out) {
  int n = (blockIdx.x * blockDim.x + threadIdx.x) >> 5;
  if (n >= NN) return;
  int lane = threadIdx.x & 31;
  const float* row = in + (size_t)n * HID;
  float v[8], s = 0.f;
#pragma unroll
  for (int i = 0; i < 8; ++i) { v[i] = row[lane + 32 * i]; s += v[i]; }
  float mean = wsum(s) * (1.0f / HID);
  float q = 0.f;
#pragma unroll
  for (int i = 0; i < 8; ++i) { float d = v[i] - mean; q += d * d; }
  float rs = rsqrtf(wsum(q) * (1.0f / HID) + 1e-5f);
#pragma unroll
  for (int i = 0; i < 8; ++i) {
    int c = lane + 32 * i;
    float y = (v[i] - mean) * rs * g[c] + b[c];
    out[(size_t)n * HID + c] = (_Float16)fmaxf(y, 0.0f);
  }
}

static inline void launch_gemm(const _Float16* A, const _Float16* Bt, const float* bias,
                               float* Cf, _Float16* Ch, int M, int K, int Nc,
                               hipStream_t s) {
  int waves = (M >> 4) * (Nc >> 6);
  int blocks = (waves + 7) / 8;
  k_gemm<<<blocks, 256, 0, s>>>(A, Bt, bias, Cf, Ch, M, K, Nc);
}

extern "C" void kernel_launch(void* const* d_in, const int* in_sizes, int n_in,
                              void* d_out, int out_size, void* d_ws, size_t ws_size,
                              hipStream_t stream) {
  (void)out_size; (void)ws_size;
  struct Ptrs {
    const float *node_emb, *type_emb;
    const float *rel_w[3], *rel_b[3], *self_w[3], *self_b[3];
    const float *ln1_g[3], *ln1_b[3], *ln2_g[3], *ln2_b[3];
    const float *qw, *qb, *kw, *kb, *vw, *vb, *ow, *ob;
    const float *w1, *b1, *lng, *lnb, *w2, *b2;
    const int *ids, *types, *esrc, *edst, *etype;
  } p;
  auto F = [&](int i) { return (const float*)d_in[i]; };
  auto I = [&](int i) { return (const int*)d_in[i]; };

  if (n_in >= 44 && in_sizes[0] == NN * EMB) {
    // insertion-order recursive flatten: params first (node_emb, type_emb, layers, attn, out)
    p.node_emb = F(0); p.type_emb = F(1);
    int idx = 2;
    for (int l = 0; l < 3; ++l) {
      p.rel_w[l] = F(idx++); p.rel_b[l] = F(idx++);
      p.self_w[l] = F(idx++); p.self_b[l] = F(idx++);
      p.ln1_g[l] = F(idx++); p.ln1_b[l] = F(idx++);
      p.ln2_g[l] = F(idx++); p.ln2_b[l] = F(idx++);
    }
    p.qw = F(idx++); p.qb = F(idx++); p.kw = F(idx++); p.kb = F(idx++);
    p.vw = F(idx++); p.vb = F(idx++); p.ow = F(idx++); p.ob = F(idx++);
    p.w1 = F(idx++); p.b1 = F(idx++); p.lng = F(idx++); p.lnb = F(idx++);
    p.w2 = F(idx++); p.b2 = F(idx++);
    p.ids = I(idx++); p.types = I(idx++);
    p.esrc = I(idx); p.edst = p.esrc + NE; idx++;
    p.etype = I(idx++);
  } else {
    // jax tree_leaves (sorted dict keys at every level)
    p.esrc = I(0); p.edst = p.esrc + NE; p.etype = I(1);
    p.ids = I(2); p.types = I(3);
    p.kb = F(4); p.kw = F(5); p.ob = F(6); p.ow = F(7);
    p.qb = F(8); p.qw = F(9); p.vb = F(10); p.vw = F(11);
    int idx = 12;
    for (int l = 0; l < 3; ++l) {
      p.ln1_b[l] = F(idx++); p.ln1_g[l] = F(idx++);
      p.ln2_b[l] = F(idx++); p.ln2_g[l] = F(idx++);
      p.rel_b[l] = F(idx++); p.rel_w[l] = F(idx++);
      p.self_b[l] = F(idx++); p.self_w[l] = F(idx++);
    }
    p.node_emb = F(idx++);
    p.b1 = F(idx++); p.b2 = F(idx++); p.lnb = F(idx++); p.lng = F(idx++);
    p.w1 = F(idx++); p.w2 = F(idx++);
    p.type_emb = F(idx++);
  }

  // ---------------- workspace carve ----------------
  char* base = (char*)d_ws;
  size_t off = 0;
  auto alloc = [&](size_t bytes) {
    void* ptr = base + off;
    off += (bytes + 255) & ~(size_t)255;
    return ptr;
  };
  float*    xA   = (float*)alloc((size_t)NN * HID * 4);          // layer input (residual)
  float*    outB = (float*)alloc((size_t)NN * HID * 4);          // conv accumulator / MLP tmp
  float*    hC   = (float*)alloc((size_t)NN * HID * 4);          // per-relation projection
  float*    L    = (float*)alloc((size_t)NN * SS * HID * 4);     // stacked layer outputs
  _Float16* x16  = (_Float16*)alloc((size_t)NN * HID * 2);       // current x in f16
  _Float16* st16 = (_Float16*)alloc((size_t)NN * SS * HID * 2);  // stacked f16
  _Float16* q16  = (_Float16*)alloc((size_t)NN * SS * HID * 2);
  _Float16* k16  = (_Float16*)alloc((size_t)NN * SS * HID * 2);
  _Float16* v16  = (_Float16*)alloc((size_t)NN * SS * HID * 2);
  _Float16* m16a = (_Float16*)alloc((size_t)NN * HID * 2);       // obar
  _Float16* m16b = (_Float16*)alloc((size_t)NN * HID * 2);       // attn-proj / mlp hidden
  const int IND[3] = {EMB, HID, HID};
  _Float16* selfw_t[3]; _Float16* relw_t[3];
  for (int l = 0; l < 3; ++l) {
    selfw_t[l] = (_Float16*)alloc((size_t)HID * IND[l] * 2);
    relw_t[l]  = (_Float16*)alloc((size_t)RR * HID * IND[l] * 2);
  }
  _Float16* qw_t = (_Float16*)alloc((size_t)HID * HID * 2);
  _Float16* kw_t = (_Float16*)alloc((size_t)HID * HID * 2);
  _Float16* vw_t = (_Float16*)alloc((size_t)HID * HID * 2);
  _Float16* ow_t = (_Float16*)alloc((size_t)HID * HID * 2);
  _Float16* w1_t = (_Float16*)alloc((size_t)HID * HID * 2);
  _Float16* w2_t = (_Float16*)alloc((size_t)EMB * HID * 2);

  // ---------------- weight prep (transpose + f16) ----------------
  for (int l = 0; l < 3; ++l) {
    int K = IND[l];
    k_wprep<<<(K * HID + 255) / 256, 256, 0, stream>>>(p.self_w[l], selfw_t[l], K, HID);
    for (int r = 0; r < RR; ++r)
      k_wprep<<<(K * HID + 255) / 256, 256, 0, stream>>>(
          p.rel_w[l] + (size_t)r * K * HID, relw_t[l] + (size_t)r * HID * K, K, HID);
  }
  k_wprep<<<(HID * HID + 255) / 256, 256, 0, stream>>>(p.qw, qw_t, HID, HID);
  k_wprep<<<(HID * HID + 255) / 256, 256, 0, stream>>>(p.kw, kw_t, HID, HID);
  k_wprep<<<(HID * HID + 255) / 256, 256, 0, stream>>>(p.vw, vw_t, HID, HID);
  k_wprep<<<(HID * HID + 255) / 256, 256, 0, stream>>>(p.ow, ow_t, HID, HID);
  k_wprep<<<(HID * HID + 255) / 256, 256, 0, stream>>>(p.w1, w1_t, HID, HID);
  k_wprep<<<(HID * EMB + 255) / 256, 256, 0, stream>>>(p.w2, w2_t, HID, EMB);

  // ---------------- embeddings ----------------
  k_gather<<<(NN * EMB + 255) / 256, 256, 0, stream>>>(p.ids, p.types, p.node_emb,
                                                       p.type_emb, x16);

  // ---------------- graph conv layers ----------------
  for (int l = 0; l < 3; ++l) {
    int K = IND[l];
    launch_gemm(x16, selfw_t[l], p.self_b[l], outB, nullptr, NN, K, HID, stream);
    for (int r = 0; r < RR; ++r) {
      launch_gemm(x16, relw_t[l] + (size_t)r * HID * K, nullptr, hC, nullptr,
                  NN, K, HID, stream);
      k_scatter<<<(NE * 32 + 255) / 256, 256, 0, stream>>>(
          p.esrc, p.edst, p.etype, hC, p.rel_b[l] + (size_t)r * HID, r, outB);
    }
    k_postconv<<<(NN * 32 + 255) / 256, 256, 0, stream>>>(
        outB, (l > 0) ? xA : nullptr, p.ln1_g[l], p.ln1_b[l], p.ln2_g[l], p.ln2_b[l],
        xA, L, x16, l);
  }

  // ---------------- attention over stacked layer outputs ----------------
  long long stcnt = (long long)NN * SS * HID;
  k_f2h<<<(unsigned)((stcnt + 255) / 256), 256, 0, stream>>>(L, st16, stcnt);
  launch_gemm(st16, qw_t, p.qb, nullptr, q16, NN * SS, HID, HID, stream);
  launch_gemm(st16, kw_t, p.kb, nullptr, k16, NN * SS, HID, HID, stream);
  launch_gemm(st16, vw_t, p.vb, nullptr, v16, NN * SS, HID, HID, stream);
  k_attn<<<(NN * NH * 32 + 255) / 256, 256, 0, stream>>>(q16, k16, v16, m16a);
  // mean over S commutes with the linear output projection
  launch_gemm(m16a, ow_t, p.ob, nullptr, m16b, NN, HID, HID, stream);

  // ---------------- output MLP ----------------
  launch_gemm(m16b, w1_t, p.b1, outB, nullptr, NN, HID, HID, stream);
  k_lnrelu<<<(NN * 32 + 255) / 256, 256, 0, stream>>>(outB, p.lng, p.lnb, m16b);
  launch_gemm(m16b, w2_t, p.b2, (float*)d_out, nullptr, NN, HID, EMB, stream);
}
